// BankedExpertsModule_57226144252168
// MI455X (gfx1250) — matrix operand order repaired and checked by reading, not compiled, not run on hardware
//
#include <hip/hip_runtime.h>

// ---------------------------------------------------------------------------
// BankedExpertsModule for MI455X (gfx1250, wave32, WMMA bf16 16x16x32)
// Shapes: N=16384 tokens, D=4096, H=2048, E=16, R=64, O=4096, top_k=2
// Workspace requirement: ~240 MB.
// GEMM1 stages W1 tiles into LDS with the Tensor Data Mover (double-buffered).
// ---------------------------------------------------------------------------

#define NN 16384
#define DD 4096
#define HH 2048
#define EE 16
#define RR 64
#define OO 4096
#define NSLOT (2 * NN)          // 32768 (token, k) slots
#define NSP 33024               // padded slots: 2N + 16*16 headroom, = 258*128 = 2064*16
#define NTILES 2064             // NSP / 16

typedef __bf16 v16bf __attribute__((ext_vector_type(16)));
typedef __bf16 v8bf  __attribute__((ext_vector_type(8)));
typedef float  v8f   __attribute__((ext_vector_type(8)));
typedef int    v4i   __attribute__((ext_vector_type(4)));
typedef int    v8i   __attribute__((ext_vector_type(8)));

// LDS B-tile row stride: 32 bf16 + 4 bf16 pad (TDM pads 2 DWORDs per 16 DWORDs)
#define BROW 36

// ---- helpers ---------------------------------------------------------------

__device__ __forceinline__ unsigned short f2bf(float f) {
    unsigned int u = __float_as_uint(f);
    u += 0x7FFFu + ((u >> 16) & 1u);          // round-to-nearest-even
    return (unsigned short)(u >> 16);
}

// gelu(tanh approx): 0.5*x*(1+tanh(z)) == x*sigmoid(2z) -- branch-free.
__device__ __forceinline__ float gelu_tanh(float x) {
    const float d = 1.f + __expf(-1.5957691216057308f * (x + 0.044715f * x * x * x));
    return x * __builtin_amdgcn_rcpf(d);
}

// A-matrix fragment, 16x32 bf16, A row-major [*, ld].
// Lanes 0-15: rows, K=k0..k0+15; lanes 16-31: K+16.
__device__ __forceinline__ v16bf frag_a(const unsigned short* __restrict__ p,
                                        int ld, int k0, int lane) {
    const int r  = lane & 15;
    const int ko = (lane >> 4) * 16;
    union { v16bf w; v8bf h[2]; } u;
    const unsigned short* q = p + (size_t)r * ld + k0 + ko;
    u.h[0] = *reinterpret_cast<const v8bf*>(q);
    u.h[1] = *reinterpret_cast<const v8bf*>(q + 8);
    return u.w;
}

// B-matrix fragment, 32x16 bf16, computing A @ Bmat^T with Bmat row-major [*, ld].
// Lanes 0-15: col n, K=k0..k0+15; lanes 16-31: K+16.
__device__ __forceinline__ v16bf frag_b(const unsigned short* __restrict__ p,
                                        int ld, int k0, int lane) {
    const int c  = lane & 15;
    const int kb = (lane >> 4) * 16;
    union { v16bf w; v8bf h[2]; } u;
    const unsigned short* q = p + (size_t)c * ld + k0 + kb;
    u.h[0] = *reinterpret_cast<const v8bf*>(q);
    u.h[1] = *reinterpret_cast<const v8bf*>(q + 8);
    return u.w;
}

// B fragment from an LDS tile (row stride BROW bf16, K within tile).
__device__ __forceinline__ v16bf frag_b_lds(const unsigned short* p, int lane) {
    const int c  = lane & 15;
    const int kb = (lane >> 4) * 16;
    union { v16bf w; v8bf h[2]; } u;
    const unsigned short* q = p + c * BROW + kb;
    u.h[0] = *reinterpret_cast<const v8bf*>(q);
    u.h[1] = *reinterpret_cast<const v8bf*>(q + 8);
    return u.w;
}

#define WMMA_BF16(A, B, C) \
    __builtin_amdgcn_wmma_f32_16x16x32_bf16(false, (A), false, (B), (short)0, (C), false, false)

// Issue one TDM 2D tile load: 128 rows x 32 bf16 cols from gbase (row stride ldElems)
// into LDS at ldsOff, padding rows to BROW bf16. Values must be wave-uniform.
__device__ __forceinline__ void tdm_load_tile(unsigned long long gaddr, unsigned ldsOff,
                                              int ldElems, int dim1) {
    v4i g0;
    g0[0] = 1;                                              // count=1 valid descriptor
    g0[1] = (int)ldsOff;                                    // lds_addr (bytes)
    g0[2] = (int)(unsigned)(gaddr & 0xFFFFFFFFull);         // global_addr[31:0]
    g0[3] = (int)((unsigned)((gaddr >> 32) & 0x01FFFFFFull) | (2u << 30)); // [56:32]|type=2
    v8i g1;
    g1[0] = (1 << 16)                                       // data_size = 2B
          | (1 << 20)                                       // pad_enable
          | (3 << 22)                                       // pad_interval: 16 DWORDs
          | (1 << 25);                                      // pad_amount: 2 DWORDs
    g1[1] = (int)((unsigned)ldElems << 16);                 // tensor_dim0[15:0] (atomic_addr=0)
    g1[2] = (int)(((unsigned)dim1 & 0xFFFFu) << 16)         // tensor_dim1[15:0]
          | (int)(((unsigned)ldElems >> 16) & 0xFFFFu);     // tensor_dim0[31:16]
    g1[3] = (32 << 16)                                      // tile_dim0 = 32 elements
          | (int)(((unsigned)dim1 >> 16) & 0xFFFFu);        // tensor_dim1[31:16]
    g1[4] = 128;                                            // tile_dim1 = 128 rows
    g1[5] = ldElems;                                        // tensor_dim0_stride[31:0]
    g1[6] = 0;                                              // stride[47:32] | dim1_stride lo
    g1[7] = 0;
    asm volatile("tensor_load_to_lds %0, %1" :: "s"(g0), "s"(g1) : "memory");
}

// ---- kernels ---------------------------------------------------------------

__global__ void k_cast(const float* __restrict__ in, unsigned short* __restrict__ out,
                       long n4) {
    long i = (long)blockIdx.x * blockDim.x + threadIdx.x;
    long stride = (long)gridDim.x * blockDim.x;
    const float4* in4 = reinterpret_cast<const float4*>(in);
    ushort4* out4 = reinterpret_cast<ushort4*>(out);
    for (; i < n4; i += stride) {
        float4 v = in4[i];
        ushort4 r;
        r.x = f2bf(v.x); r.y = f2bf(v.y); r.z = f2bf(v.z); r.w = f2bf(v.w);
        out4[i] = r;
    }
}

// GEMM1: hg = gelu(x @ W1^T + b1). Block tile 128(M) x 128(H), 8 waves,
// each wave 32x64 = 2x4 WMMA accumulators, K step 32.
// W1 tile (128x32) staged into LDS by the Tensor Data Mover, double-buffered.
__global__ __launch_bounds__(256) void k_gemm1(const unsigned short* __restrict__ xb,
                                               const unsigned short* __restrict__ w1b,
                                               const float* __restrict__ b1,
                                               unsigned short* __restrict__ hgb) {
    __shared__ unsigned short bsm[2][128][BROW];
    const int lane = threadIdx.x & 31, wave = threadIdx.x >> 5;
    const int m0 = blockIdx.y * 128 + (wave >> 1) * 32;
    const int hw0 = (wave & 1) * 64;                       // wave's H offset inside block
    const int h0 = blockIdx.x * 128 + hw0;

    const unsigned long long gbase =
        (unsigned long long)(uintptr_t)w1b + ((unsigned long long)(blockIdx.x * 128) * DD) * 2ull;

    v8f acc[2][4] = {};
    if (wave == 0)
        tdm_load_tile(gbase, (unsigned)(uintptr_t)&bsm[0][0][0], DD, HH);

    int p = 0;
    for (int k0 = 0; k0 < DD; k0 += 32) {
        if (wave == 0) __builtin_amdgcn_s_wait_tensorcnt(0);
        __syncthreads();                                   // tile p ready; buffer p^1 free
        if (wave == 0 && k0 + 32 < DD)
            tdm_load_tile(gbase + (unsigned long long)(k0 + 32) * 2ull,
                          (unsigned)(uintptr_t)&bsm[p ^ 1][0][0], DD, HH);

        v16bf a[2], b[4];
#pragma unroll
        for (int i = 0; i < 2; ++i)
            a[i] = frag_a(xb + (size_t)(m0 + i * 16) * DD, DD, k0, lane);
#pragma unroll
        for (int j = 0; j < 4; ++j)
            b[j] = frag_b_lds(&bsm[p][hw0 + j * 16][0], lane);
#pragma unroll
        for (int i = 0; i < 2; ++i)
#pragma unroll
            for (int j = 0; j < 4; ++j)
                acc[i][j] = WMMA_BF16(a[i], b[j], acc[i][j]);
        p ^= 1;
    }

    const int cr = (lane >> 4) * 8, cc = lane & 15;
#pragma unroll
    for (int i = 0; i < 2; ++i)
#pragma unroll
        for (int j = 0; j < 4; ++j) {
            const int col = h0 + j * 16 + cc;
            const float bias = b1[col];
#pragma unroll
            for (int v = 0; v < 8; ++v) {
                const int row = m0 + i * 16 + cr + v;
                hgb[(size_t)row * HH + col] = f2bf(gelu_tanh(acc[i][j][v] + bias));
            }
        }
}

// Gating: logits = hg @ W2^T + b2 (one 16x16 WMMA tile per wave over K=2048),
// then softmax (aux-loss partials), top-2 + 2-way softmax, expert counts.
__global__ __launch_bounds__(256) void k_gate(const unsigned short* __restrict__ hgb,
                                              const unsigned short* __restrict__ w2b,
                                              const float* __restrict__ b2,
                                              float* __restrict__ probsum,
                                              int* __restrict__ counts,
                                              int* __restrict__ topI,
                                              float* __restrict__ topG) {
    __shared__ float sl[8][16][16];
    __shared__ float ps[16];
    const int lane = threadIdx.x & 31, wave = threadIdx.x >> 5;
    const int row0 = blockIdx.x * 128 + wave * 16;
    if (threadIdx.x < 16) ps[threadIdx.x] = 0.f;

    v8f acc = {};
    for (int k0 = 0; k0 < HH; k0 += 32) {
        v16bf a = frag_a(hgb + (size_t)row0 * HH, HH, k0, lane);
        v16bf b = frag_b(w2b, HH, k0, lane);
        acc = WMMA_BF16(a, b, acc);
    }
#pragma unroll
    for (int v = 0; v < 8; ++v)
        sl[wave][(lane >> 4) * 8 + v][lane & 15] = acc[v] + b2[lane & 15];
    __syncthreads();

    if (threadIdx.x < 128) {
        const int wv = threadIdx.x >> 4, rl = threadIdx.x & 15;
        const int row = blockIdx.x * 128 + wv * 16 + rl;
        float* lg = sl[wv][rl];
        float m = lg[0];
#pragma unroll
        for (int e = 1; e < EE; ++e) m = fmaxf(m, lg[e]);
        float s = 0.f, pr[EE];
#pragma unroll
        for (int e = 0; e < EE; ++e) { pr[e] = __expf(lg[e] - m); s += pr[e]; }
        const float inv = 1.f / s;
#pragma unroll
        for (int e = 0; e < EE; ++e) atomicAdd(&ps[e], pr[e] * inv);
        // top-2 (ties -> lowest index, matching lax.top_k)
        int e0 = 0; float v0 = lg[0];
#pragma unroll
        for (int e = 1; e < EE; ++e) if (lg[e] > v0) { v0 = lg[e]; e0 = e; }
        int e1 = (e0 == 0) ? 1 : 0; float v1 = lg[e1];
#pragma unroll
        for (int e = 0; e < EE; ++e)
            if (e != e0 && lg[e] > v1) { v1 = lg[e]; e1 = e; }
        const float g0 = 1.f / (1.f + __expf(v1 - v0));
        topI[row * 2] = e0; topI[row * 2 + 1] = e1;
        topG[row * 2] = g0; topG[row * 2 + 1] = 1.f - g0;
        atomicAdd(&counts[e0], 1);
        atomicAdd(&counts[e1], 1);
    }
    __syncthreads();
    if (threadIdx.x < 16) atomicAdd(&probsum[threadIdx.x], ps[threadIdx.x]);
}

// Serial scan: 16-row-padded per-expert offsets + tile->expert map.
__global__ void k_scan(const int* __restrict__ counts, int* __restrict__ cursor,
                       int* __restrict__ tileExpert) {
    if (threadIdx.x == 0) {
        int off = 0;
        for (int e = 0; e < EE; ++e) {
            cursor[e] = off;
            const int t = (counts[e] + 15) >> 4;
            for (int tt = 0; tt < t; ++tt) tileExpert[(off >> 4) + tt] = e;
            off += t * 16;
        }
        for (int t = off >> 4; t < NTILES; ++t) tileExpert[t] = 0;  // inert tiles
    }
}

__global__ void k_scatter(const int* __restrict__ topI, const float* __restrict__ topG,
                          int* __restrict__ cursor, int* __restrict__ rowmap,
                          float* __restrict__ gateval) {
    const int idx = blockIdx.x * 256 + threadIdx.x;
    if (idx < NSLOT) {
        const int e = topI[idx];
        const int pos = atomicAdd(&cursor[e], 1);
        rowmap[pos] = idx >> 1;
        gateval[pos] = topG[idx];
    }
}

// A-projection (gathered rows) + LayerNorm(R)*gamma+beta, *scaling*gate -> bf16 hw.
__global__ __launch_bounds__(256) void k_aproj(const unsigned short* __restrict__ xb,
                                               const unsigned short* __restrict__ Ab,
                                               const float* __restrict__ scaling,
                                               const float* __restrict__ gamma,
                                               const float* __restrict__ beta,
                                               const int* __restrict__ rowmap,
                                               const float* __restrict__ gateval,
                                               const int* __restrict__ tileExpert,
                                               unsigned short* __restrict__ hwb) {
    __shared__ float sh[8][16][RR];
    const int lane = threadIdx.x & 31, wave = threadIdx.x >> 5;
    const int slot0 = blockIdx.x * 128 + wave * 16;
    const int e = tileExpert[slot0 >> 4];
    const unsigned short* Ae = Ab + (size_t)e * RR * DD;

    const int n = rowmap[slot0 + (lane & 15)];            // gathered token row
    const unsigned short* arow = xb + (size_t)n * DD;
    const int kof = (lane >> 4) * 16;

    v8f acc[4] = {};
    for (int k0 = 0; k0 < DD; k0 += 32) {
        union { v16bf w; v8bf h[2]; } ua;
        const unsigned short* q = arow + k0 + kof;
        ua.h[0] = *reinterpret_cast<const v8bf*>(q);
        ua.h[1] = *reinterpret_cast<const v8bf*>(q + 8);
        v16bf b[4];
#pragma unroll
        for (int j = 0; j < 4; ++j)
            b[j] = frag_b(Ae + (size_t)(j * 16) * DD, DD, k0, lane);
#pragma unroll
        for (int j = 0; j < 4; ++j)
            acc[j] = WMMA_BF16(ua.w, b[j], acc[j]);
    }
#pragma unroll
    for (int j = 0; j < 4; ++j)
#pragma unroll
        for (int v = 0; v < 8; ++v)
            sh[wave][(lane >> 4) * 8 + v][j * 16 + (lane & 15)] = acc[j][v];
    __syncthreads();

    if (threadIdx.x < 128) {
        const int wv = threadIdx.x >> 4, rl = threadIdx.x & 15;
        const int slot = blockIdx.x * 128 + wv * 16 + rl;
        const int ee = tileExpert[slot >> 4];
        float* hr = sh[wv][rl];
        float mu = 0.f;
#pragma unroll
        for (int r = 0; r < RR; ++r) mu += hr[r];
        mu *= (1.f / RR);
        float var = 0.f;
#pragma unroll
        for (int r = 0; r < RR; ++r) { const float d = hr[r] - mu; var += d * d; }
        var *= (1.f / RR);
        const float rstd = rsqrtf(var + 1e-5f);
        const float sg = scaling[ee] * gateval[slot];     // gate==0 on padded slots
#pragma unroll
        for (int r = 0; r < RR; ++r) {
            const float hn = (hr[r] - mu) * rstd * gamma[ee * RR + r] + beta[ee * RR + r];
            hwb[(size_t)slot * RR + r] = f2bf(hn * sg);
        }
    }
}

// B-projection: y[n,:] += hw[slot,:] @ B[e]^T ; K=64 -> 2 WMMA k-steps; atomic scatter.
__global__ __launch_bounds__(256) void k_bproj(const unsigned short* __restrict__ hwb,
                                               const unsigned short* __restrict__ Bb,
                                               const int* __restrict__ rowmap,
                                               const int* __restrict__ tileExpert,
                                               float* __restrict__ y) {
    const int lane = threadIdx.x & 31, wave = threadIdx.x >> 5;
    const int tile = blockIdx.y;
    const int e = tileExpert[tile];
    const int slot0 = tile * 16;
    const int o0 = blockIdx.x * 256 + wave * 32;
    const unsigned short* Be = Bb + (size_t)e * OO * RR;

    v8f acc[2] = {};
#pragma unroll
    for (int k0 = 0; k0 < RR; k0 += 32) {
        v16bf a = frag_a(hwb + (size_t)slot0 * RR, RR, k0, lane);
        v16bf b0 = frag_b(Be + (size_t)o0 * RR, RR, k0, lane);
        v16bf b1 = frag_b(Be + (size_t)(o0 + 16) * RR, RR, k0, lane);
        acc[0] = WMMA_BF16(a, b0, acc[0]);
        acc[1] = WMMA_BF16(a, b1, acc[1]);
    }
    const int cr = (lane >> 4) * 8, cc = lane & 15;
#pragma unroll
    for (int t = 0; t < 2; ++t)
#pragma unroll
        for (int v = 0; v < 8; ++v) {
            const int slot = slot0 + cr + v;
            const int n = rowmap[slot];                   // padded -> row 0, acc==0
            unsafeAtomicAdd(&y[(size_t)n * OO + o0 + t * 16 + cc], acc[t][v]);
        }
}

__global__ void k_loss(const float* __restrict__ probsum, float* __restrict__ out) {
    __shared__ float s[EE];
    const int t = threadIdx.x;
    if (t < EE) {
        const float p = probsum[t] * (1.f / (float)NN);
        const float d = p - 1.f / (float)EE;
        s[t] = d * d;
    }
    __syncthreads();
    if (t == 0) {
        float a = 0.f;
        for (int e = 0; e < EE; ++e) a += s[e];
        out[0] = a * (1.f / (float)EE);
    }
}

// ---- host ------------------------------------------------------------------

extern "C" void kernel_launch(void* const* d_in, const int* in_sizes, int n_in,
                              void* d_out, int out_size, void* d_ws, size_t ws_size,
                              hipStream_t stream) {
    const float* x    = (const float*)d_in[0];
    const float* w1   = (const float*)d_in[1];
    const float* b1   = (const float*)d_in[2];
    const float* w2   = (const float*)d_in[3];
    const float* b2   = (const float*)d_in[4];
    const float* A    = (const float*)d_in[5];
    const float* B    = (const float*)d_in[6];
    const float* scal = (const float*)d_in[7];
    const float* gmm  = (const float*)d_in[8];
    const float* bta  = (const float*)d_in[9];
    // top_k assumed == 2 (in_sizes[10] == 1)
    float* y = (float*)d_out;

    char* ws = (char*)d_ws;
    size_t o = 0;
    auto alloc = [&](size_t bytes) -> size_t {
        size_t cur = (o + 255) & ~(size_t)255;
        o = cur + bytes;
        return cur;
    };
    unsigned short* xb  = (unsigned short*)(ws + alloc((size_t)NN * DD * 2));
    unsigned short* w1b = (unsigned short*)(ws + alloc((size_t)HH * DD * 2));
    unsigned short* w2b = (unsigned short*)(ws + alloc((size_t)EE * HH * 2));
    unsigned short* Ab  = (unsigned short*)(ws + alloc((size_t)EE * RR * DD * 2));
    unsigned short* Bb  = (unsigned short*)(ws + alloc((size_t)EE * OO * RR * 2));
    unsigned short* hgb = (unsigned short*)(ws + alloc((size_t)NN * HH * 2));
    unsigned short* hwb = (unsigned short*)(ws + alloc((size_t)NSP * RR * 2));
    int*   topI = (int*)(ws + alloc((size_t)NSLOT * 4));
    float* topG = (float*)(ws + alloc((size_t)NSLOT * 4));
    // zero-initialized meta region
    size_t metaStart = (o + 255) & ~(size_t)255;
    int*   counts     = (int*)(ws + alloc(EE * 4));
    int*   cursor     = (int*)(ws + alloc(EE * 4));
    float* probsum    = (float*)(ws + alloc(EE * 4));
    int*   tileExpert = (int*)(ws + alloc((size_t)NTILES * 4));
    int*   rowmap     = (int*)(ws + alloc((size_t)NSP * 4));
    float* gateval    = (float*)(ws + alloc((size_t)NSP * 4));
    size_t metaEnd = o;

    hipMemsetAsync(ws + metaStart, 0, metaEnd - metaStart, stream);
    hipMemsetAsync(d_out, 0, (size_t)out_size * sizeof(float), stream);

    // bf16 conversions
    k_cast<<<2048, 256, 0, stream>>>(x,  xb,  (long)NN * DD / 4);
    k_cast<<<1024, 256, 0, stream>>>(w1, w1b, (long)HH * DD / 4);
    k_cast<<<32,   256, 0, stream>>>(w2, w2b, (long)EE * HH / 4);
    k_cast<<<512,  256, 0, stream>>>(A,  Ab,  (long)EE * RR * DD / 4);
    k_cast<<<512,  256, 0, stream>>>(B,  Bb,  (long)EE * OO * RR / 4);

    // gating network
    k_gemm1<<<dim3(HH / 128, NN / 128), 256, 0, stream>>>(xb, w1b, b1, hgb);
    k_gate<<<NN / 128, 256, 0, stream>>>(hgb, w2b, b2, probsum, counts, topI, topG);
    k_scan<<<1, 32, 0, stream>>>(counts, cursor, tileExpert);
    k_scatter<<<(NSLOT + 255) / 256, 256, 0, stream>>>(topI, topG, cursor, rowmap, gateval);

    // sparse LoRA experts (top-2 only)
    k_aproj<<<NSP / 128, 256, 0, stream>>>(xb, Ab, scal, gmm, bta, rowmap, gateval,
                                           tileExpert, hwb);
    k_bproj<<<dim3(OO / 256, NTILES), 256, 0, stream>>>(hwb, Bb, rowmap, tileExpert, y);

    // aux load-balancing loss
    k_loss<<<1, 32, 0, stream>>>(probsum, y + (size_t)NN * OO);
}
